// MaskedGraphAttention_14053132993204
// MI455X (gfx1250) — matrix operand order repaired and checked by reading, not compiled, not run on hardware
//
#include <hip/hip_runtime.h>
#include <hip/hip_bf16.h>

typedef __bf16 bf;
typedef __attribute__((ext_vector_type(16))) __bf16 v16bf;
typedef __attribute__((ext_vector_type(8)))  __bf16 v8bf;
typedef __attribute__((ext_vector_type(8)))  float  v8f;
typedef __attribute__((ext_vector_type(4)))  float  v4f;

#define J_DIM 22
#define D_DIM 128
#define H_DIM 8
#define N_INST (64*240)
#define NW 8                 // waves per block (wave32 -> 256 threads)
#define PW 17408             // per-wave LDS bytes

__device__ __forceinline__ v16bf cat8(v8bf lo, v8bf hi) {
  return __builtin_shufflevector(lo, hi, 0,1,2,3,4,5,6,7,8,9,10,11,12,13,14,15);
}

__device__ __forceinline__ v8f wmma_bf16(v16bf a, v16bf b, v8f c) {
  // D = A(16x32 bf16) * B(32x16 bf16) + C(16x16 f32)
  return __builtin_amdgcn_wmma_f32_16x16x32_bf16(false, a, false, b, (short)0, c, false, false);
}

// B-fragment loader from a row-major bf16 matrix W[rows][128]:
// lane holds column o = ntile*16 + (lane&15), K = (lane>>4)*8 .. +7 and +16 .. +23.
__device__ __forceinline__ v16bf load_wfrag(const bf* __restrict__ W, int ntile, int kc, int lane) {
  const int o  = ntile*16 + (lane & 15);
  const int d0 = kc*32 + (lane >> 4)*8;
  const v8bf* p = (const v8bf*)(W + o*D_DIM + d0);
  return cat8(p[0], p[2]);
}

__global__ void cvt_weights(const float* __restrict__ wq, const float* __restrict__ wo,
                            bf* __restrict__ dst) {
  int i = blockIdx.x * blockDim.x + threadIdx.x;
  if (i < 384*128) dst[i] = (bf)wq[i];
  if (i < 128*128) dst[384*128 + i] = (bf)wo[i];
}

__global__ __launch_bounds__(NW*32, 1)
void mga_kernel(const float* __restrict__ x, const int* __restrict__ adj,
                const float* __restrict__ bqkv, const float* __restrict__ bout,
                const bf* __restrict__ Wqkv, const bf* __restrict__ Wout,
                float* __restrict__ out)
{
  extern __shared__ char smem[];
  const int lane = threadIdx.x & 31;
  const int wave = threadIdx.x >> 5;
  const int inst = blockIdx.x * NW + wave;
  if (inst >= N_INST) return;     // wave-uniform

  char* base = smem + wave * PW;
  bf*    Qs = (bf*)   (base);          // [32 j][16 dd]   j-major   (1 KB)
  bf*    Ks = (bf*)   (base + 1024);   // [32 j][16 dd]   j-major   (1 KB)
  bf*    Vt = (bf*)   (base + 2048);   // [16 dd][32 j]   dd-major  (1 KB)
  float* Ss = (float*)(base + 3072);   // [32 j][32 k]    scores f32 (4 KB)
  bf*    Ps = (bf*)   (base + 7168);   // [32 j][32 k]    probs bf16 (2 KB)
  bf*    Ys = (bf*)   (base + 9216);   // [32 j][128 d]   y bf16     (8 KB)

  const int row = lane & 15;
  const int kh  = lane >> 4;           // K-half select per ISA A/B layouts

  // --- adjacency bitmask for this lane's query row (reused across heads)
  unsigned maskbits = 0u;
  if (lane < J_DIM) {
    const int* arow = adj + lane * J_DIM;
    #pragma unroll
    for (int k = 0; k < J_DIM; ++k)
      if (arow[k] > 0) maskbits |= (1u << k);
  }

  // --- preload X as bf16 A-fragments: 2 M-tiles x 4 K-chunks (64 VGPRs)
  const float* xi = x + (size_t)inst * (J_DIM * D_DIM);
  v16bf ax[2][4];
  #pragma unroll
  for (int mt = 0; mt < 2; ++mt) {
    const int j = mt*16 + row;
    #pragma unroll
    for (int kc = 0; kc < 4; ++kc) {
      union { v16bf v; bf e[16]; } a;
      if (j < J_DIM) {
        const float* xr = xi + j*D_DIM + kc*32 + kh*8;
        v4f f0 = *(const v4f*)(xr);
        v4f f1 = *(const v4f*)(xr + 4);
        v4f f2 = *(const v4f*)(xr + 16);
        v4f f3 = *(const v4f*)(xr + 20);
        #pragma unroll
        for (int e = 0; e < 4; ++e) {
          a.e[e]    = (bf)f0[e];
          a.e[4+e]  = (bf)f1[e];
          a.e[8+e]  = (bf)f2[e];
          a.e[12+e] = (bf)f3[e];
        }
      } else {
        #pragma unroll
        for (int e = 0; e < 16; ++e) a.e[e] = (bf)0.f;
      }
      ax[mt][kc] = a.v;
    }
  }

  // =================== per-head attention ===================
  #pragma unroll 1
  for (int h = 0; h < H_DIM; ++h) {
    // --- QKV projections for this head (N-tile = m3*8 + h)
    #pragma unroll
    for (int m3 = 0; m3 < 3; ++m3) {
      const int ntile = m3*8 + h;
      const float bias = bqkv[ntile*16 + row];
      #pragma unroll
      for (int mt = 0; mt < 2; ++mt) {
        v8f acc = {};
        #pragma unroll
        for (int kc = 0; kc < 4; ++kc)
          acc = wmma_bf16(ax[mt][kc], load_wfrag(Wqkv, ntile, kc, lane), acc);
        const int jb = mt*16 + kh*8;
        if (m3 == 2) {          // V -> dd-major (contiguous 8 per lane)
          #pragma unroll
          for (int r = 0; r < 8; ++r)
            Vt[row*32 + jb + r] = (bf)(acc[r] + bias);
        } else {                // Q/K -> j-major
          bf* T = (m3 == 0) ? Qs : Ks;
          #pragma unroll
          for (int r = 0; r < 8; ++r)
            T[(jb + r)*16 + row] = (bf)(acc[r] + bias);
        }
      }
    }

    // --- scores = (Q K^T) * 1/sqrt(16), K-dim = 16 (upper half zero)
    #pragma unroll
    for (int mt = 0; mt < 2; ++mt) {
      union { v16bf v; bf e[16]; v8bf h2[2]; } aq;
      #pragma unroll
      for (int e = 0; e < 16; ++e) aq.e[e] = (bf)0.f;
      aq.h2[0] = *(const v8bf*)(Qs + (mt*16 + row)*16 + kh*8);
      #pragma unroll
      for (int nt = 0; nt < 2; ++nt) {
        union { v16bf v; bf e[16]; v8bf h2[2]; } bk;
        #pragma unroll
        for (int e = 0; e < 16; ++e) bk.e[e] = (bf)0.f;
        {
          const v8bf* p = (const v8bf*)(Ks + (nt*16 + row)*16 + kh*8);
          bk.h2[0] = p[0];
        }
        v8f s = {};
        s = wmma_bf16(aq.v, bk.v, s);
        #pragma unroll
        for (int r = 0; r < 8; ++r)
          Ss[(mt*16 + kh*8 + r)*32 + nt*16 + row] = 0.25f * s[r];
      }
    }

    // --- masked softmax: lane j owns query row j
    {
      bf* prow = Ps + lane*32;
      if (lane < J_DIM) {
        const float* srow = Ss + lane*32;
        float mx = -1e30f;
        #pragma unroll
        for (int k = 0; k < J_DIM; ++k)
          if ((maskbits >> k) & 1u) mx = fmaxf(mx, srow[k]);
        float ev[J_DIM];
        float sum = 0.f;
        #pragma unroll
        for (int k = 0; k < J_DIM; ++k) {
          float e = ((maskbits >> k) & 1u) ? __expf(srow[k] - mx) : 0.f;
          ev[k] = e;
          sum += e;
        }
        const float inv = 1.f / sum;
        #pragma unroll
        for (int k = 0; k < J_DIM; ++k) prow[k] = (bf)(ev[k] * inv);
        #pragma unroll
        for (int k = J_DIM; k < 32; ++k) prow[k] = (bf)0.f;
      } else {
        #pragma unroll
        for (int k = 0; k < 32; ++k) prow[k] = (bf)0.f;
      }
    }

    // --- y_h = P(22x32pad) @ V(32x16), K-dim = 32 (zero-padded probs)
    #pragma unroll
    for (int mt = 0; mt < 2; ++mt) {
      const v8bf* pp = (const v8bf*)(Ps + (mt*16 + row)*32 + kh*8);
      v16bf ap = cat8(pp[0], pp[2]);                 // K=k0..k0+7, k0+16..k0+23
      const v8bf* pv = (const v8bf*)(Vt + row*32 + kh*8);
      v16bf bv = cat8(pv[0], pv[2]);                 // col dd=row, K=k0..k0+7, k0+16..k0+23
      v8f y = {};
      y = wmma_bf16(ap, bv, y);
      #pragma unroll
      for (int r = 0; r < 8; ++r)
        Ys[(mt*16 + kh*8 + r)*D_DIM + h*16 + row] = (bf)y[r];
    }
  }

  // =================== output projection: Y[22,128] @ Wout^T ===================
  v16bf ay[2][4];
  #pragma unroll
  for (int mt = 0; mt < 2; ++mt) {
    #pragma unroll
    for (int kc = 0; kc < 4; ++kc) {
      const v8bf* p = (const v8bf*)(Ys + (mt*16 + row)*D_DIM + kc*32 + kh*8);
      ay[mt][kc] = cat8(p[0], p[2]);
    }
  }

  float* op = out + (size_t)inst * (J_DIM * D_DIM);
  #pragma unroll 1
  for (int nt = 0; nt < 8; ++nt) {
    const float bias = bout[nt*16 + row];
    #pragma unroll
    for (int mt = 0; mt < 2; ++mt) {
      v8f acc = {};
      #pragma unroll
      for (int kc = 0; kc < 4; ++kc)
        acc = wmma_bf16(ay[mt][kc], load_wfrag(Wout, nt, kc, lane), acc);
      #pragma unroll
      for (int r = 0; r < 8; ++r) {
        const int j = mt*16 + kh*8 + r;
        if (j < J_DIM) op[j*D_DIM + nt*16 + row] = acc[r] + bias;
      }
    }
  }
}

extern "C" void kernel_launch(void* const* d_in, const int* in_sizes, int n_in,
                              void* d_out, int out_size, void* d_ws, size_t ws_size,
                              hipStream_t stream) {
  (void)in_sizes; (void)n_in; (void)out_size; (void)ws_size;
  const float* x   = (const float*)d_in[0];
  const int*   adj = (const int*)d_in[1];
  const float* wq  = (const float*)d_in[2];
  const float* bq  = (const float*)d_in[3];
  const float* wo  = (const float*)d_in[4];
  const float* bo  = (const float*)d_in[5];
  float* outp = (float*)d_out;
  bf* wbf = (bf*)d_ws;                       // [384*128] Wqkv bf16, then [128*128] Wout bf16

  hipLaunchKernelGGL(cvt_weights, dim3((384*128 + 255)/256), dim3(256), 0, stream,
                     wq, wo, wbf);

  const int blocks = (N_INST + NW - 1) / NW;  // 1920 blocks, 1 instance per wave
  const size_t shmem = (size_t)PW * NW;       // 136 KB dynamic LDS (<< 320 KB/WGP)
  hipLaunchKernelGGL(mga_kernel, dim3(blocks), dim3(NW*32), shmem, stream,
                     x, adj, bq, bo, wbf, wbf + 384*128, outp);
}
